// ActorCritic_49624052138423
// MI455X (gfx1250) — matrix operand order repaired and checked by reading, not compiled, not run on hardware
//
#include <hip/hip_runtime.h>
#include <math.h>

typedef __attribute__((ext_vector_type(16))) _Float16 v16h;
typedef __attribute__((ext_vector_type(8)))  float    v8f;

#define N_NODES 225
#define BATCH   8
#define TOTAL   (BATCH * N_NODES)      // 1800
#define MPAD    1808                   // 113 tiles * 16 rows
#define HID     64
#define E_CNT   14400
#define NPAIR   (N_NODES * N_NODES)    // 50625
#define BN_EPS  1e-5f

// ---------------------------------------------------------------- utilities
__global__ void k_copy(const float* __restrict__ src, float* __restrict__ dst, int n) {
    int i = blockIdx.x * blockDim.x + threadIdx.x;
    if (i < n) dst[i] = src[i];
}

// segment_sum(h[src], dst) accumulated on top of pre-copied self features
__global__ void k_edge_scatter(const float* __restrict__ h,
                               const int* __restrict__ srcI,
                               const int* __restrict__ dstI,
                               float* __restrict__ acc, int feat) {
    int i = blockIdx.x * blockDim.x + threadIdx.x;
    if (i >= E_CNT * feat) return;
    int e = i / feat, f = i - e * feat;
    atomicAdd(&acc[dstI[e] * feat + f], h[srcI[e] * feat + f]);
}

// first GIN linear: (1800x2) @ (2x64) + b  — K=2, scalar FMA is optimal
__global__ void k_lin_in(const float* __restrict__ a0, const float* __restrict__ w,
                         const float* __restrict__ b, float* __restrict__ h1) {
    int i = blockIdx.x * blockDim.x + threadIdx.x;
    if (i >= TOTAL * HID) return;
    int n = i >> 6, f = i & 63;
    h1[i] = a0[n * 2] * w[f] + a0[n * 2 + 1] * w[HID + f] + b[f];
}

// BatchNorm train-stats folded to per-feature scale/shift: y = x*scale + shift
__global__ void k_bn_stats(const float* __restrict__ h,
                           const float* __restrict__ gamma,
                           const float* __restrict__ beta,
                           float* __restrict__ scale, float* __restrict__ shift) {
    __shared__ float s1[256], s2[256];
    int f = blockIdx.x, t = threadIdx.x;
    float sum = 0.f, sq = 0.f;
    for (int n = t; n < TOTAL; n += 256) {
        float v = h[n * HID + f];
        sum += v; sq += v * v;
    }
    s1[t] = sum; s2[t] = sq; __syncthreads();
    for (int s = 128; s > 0; s >>= 1) {
        if (t < s) { s1[t] += s1[t + s]; s2[t] += s2[t + s]; }
        __syncthreads();
    }
    if (t == 0) {
        float m = s1[0] * (1.0f / TOTAL);
        float v = s2[0] * (1.0f / TOTAL) - m * m;
        float inv = rsqrtf(v + BN_EPS);
        float sc = gamma[f] * inv;
        scale[f] = sc;
        shift[f] = beta[f] - m * sc;
    }
}

// Pack a 64 x N f32 weight matrix (row-major, ld = N... passed ldb) into WMMA
// B-fragment order as f16. Fragment layout: for tile nt, K-step kb, lane l:
// 16 contiguous halves, element s = B[(kb*32 + (s>>3)*16 + (l>>4)*8 + (s&7))*ldb + nt*16 + (l&15)]
__global__ void k_pack_b(const float* __restrict__ B, int ldb, int N,
                         _Float16* __restrict__ Bpack) {
    int i = blockIdx.x * blockDim.x + threadIdx.x;
    int tot = (N / 16) * 2 * 32 * 16;
    if (i >= tot) return;
    int s    = i & 15;
    int lane = (i >> 4) & 31;
    int kb   = (i >> 9) & 1;
    int nt   = i >> 10;
    int k    = kb * 32 + ((s >> 3) << 4) + ((lane >> 4) << 3) + (s & 7);
    Bpack[i] = (_Float16)B[k * ldb + nt * 16 + (lane & 15)];
}

template<int MODE>  // 0: identity, 1: BN scale/shift + relu
__device__ inline float4 xform(float4 v, const float4* sc4, const float4* sh4, int idx) {
    if (MODE) {
        float4 s = sc4[idx], h = sh4[idx];
        v.x = fmaxf(v.x * s.x + h.x, 0.f);
        v.y = fmaxf(v.y * s.y + h.y, 0.f);
        v.z = fmaxf(v.z * s.z + h.z, 0.f);
        v.w = fmaxf(v.w * s.w + h.w, 0.f);
    }
    return v;
}

#define CVT4(dst, off, f4) \
    { dst[(off)] = (_Float16)(f4).x; dst[(off)+1] = (_Float16)(f4).y; \
      dst[(off)+2] = (_Float16)(f4).z; dst[(off)+3] = (_Float16)(f4).w; }

// out[M x N] = T(A[MPAD x 64]) @ B[64 x N] + bias ; one wave per 16x16 tile.
// A rows >= M are pad rows (garbage fine: never stored). No branches before WMMA.
template<int MODE>
__global__ void k_gemm64_wmma(const float* __restrict__ A, int M,
                              const float* __restrict__ scale,
                              const float* __restrict__ shift,
                              const _Float16* __restrict__ Bpack,
                              const float* __restrict__ bias,
                              float* __restrict__ out, int N) {
    int lane = threadIdx.x & 31;
    int wave = threadIdx.x >> 5;          // = N tile
    int row0 = blockIdx.x * 16;
    int half = lane >> 4;
    int r    = lane & 15;
    const float4* arow4 = (const float4*)(A + (row0 + r) * HID);
    const float4* sc4   = (const float4*)scale;
    const float4* sh4   = (const float4*)shift;
    const v16h*   bp    = (const v16h*)Bpack + wave * 64;   // 2 K-steps * 32 lanes

    v8f acc = {};
#pragma unroll
    for (int kb = 0; kb < 2; ++kb) {
        int i0 = kb * 8 + half * 2;       // float4 index of run1 (k = kb*32 + half*8)
        float4 p0 = xform<MODE>(arow4[i0    ], sc4, sh4, i0    );
        float4 p1 = xform<MODE>(arow4[i0 + 1], sc4, sh4, i0 + 1);
        float4 p2 = xform<MODE>(arow4[i0 + 4], sc4, sh4, i0 + 4);  // run2 = +16 k
        float4 p3 = xform<MODE>(arow4[i0 + 5], sc4, sh4, i0 + 5);
        v16h af;
        CVT4(af, 0, p0); CVT4(af, 4, p1); CVT4(af, 8, p2); CVT4(af, 12, p3);
        v16h bf = bp[kb * 32 + lane];     // pre-packed B fragment (2x b128)
        acc = __builtin_amdgcn_wmma_f32_16x16x32_f16(false, af, false, bf,
                                                     (short)0, acc, false, false);
    }
    int n = wave * 16 + r;
    float bv = bias ? bias[n] : 0.0f;
#pragma unroll
    for (int i = 0; i < 8; ++i) {
        int m = row0 + i + 8 * half;      // D layout: VGPR i -> M = i + 8*half
        if (m < M) out[m * N + n] = acc[i] + bv;
    }
}

// mean graph pooling: emb[b][f] = mean_n feats[b,n,f]   (512 threads, 1 block)
__global__ void k_pool(const float* __restrict__ feats, float* __restrict__ emb) {
    int i = threadIdx.x;
    int b = i >> 6, f = i & 63;
    float s = 0.f;
    for (int n = 0; n < N_NODES; ++n) s += feats[(b * N_NODES + n) * HID + f];
    emb[i] = s * (1.0f / N_NODES);
}

// critic MLP: value[b] = relu(emb @ c_w1 + b1) @ c_w2 + b2
__global__ void k_critic(const float* __restrict__ emb,
                         const float* __restrict__ w1, const float* __restrict__ b1,
                         const float* __restrict__ w2, const float* __restrict__ b2,
                         float* __restrict__ val_out) {
    int b = threadIdx.x;
    if (b >= BATCH) return;
    float v = b2[0];
    for (int h = 0; h < 32; ++h) {
        float a = b1[h];
        for (int f = 0; f < HID; ++f) a += emb[b * HID + f] * w1[f * 32 + h];
        v += fmaxf(a, 0.0f) * w2[h];
    }
    val_out[b] = v;
}

// S[b][j] = emb[b] @ a_w1[0:64] + a_b1   (actor bias folded in here once)
__global__ void k_state_proj(const float* __restrict__ emb,
                             const float* __restrict__ aw1,
                             const float* __restrict__ ab1, float* __restrict__ S) {
    int i = threadIdx.x;          // 256 = 8*32
    int b = i >> 5, j = i & 31;
    float s = ab1[j];
    for (int f = 0; f < HID; ++f) s += emb[b * HID + f] * aw1[f * 32 + j];
    S[i] = s;
}

// logits[b, i*225+j] = relu(S[b] + U[b][j] + V[b][i]) . a_w2 + a_b2
__global__ void k_logits(const float* __restrict__ S, const float* __restrict__ U,
                         const float* __restrict__ V, const float* __restrict__ aw2,
                         const float* __restrict__ ab2, float* __restrict__ out) {
    int idx = blockIdx.x * blockDim.x + threadIdx.x;
    if (idx >= BATCH * NPAIR) return;
    int b = idx / NPAIR, p = idx - b * NPAIR;
    int i = p / N_NODES, j = p - i * N_NODES;   // nodes1 = tile (j inner), nodes2 = repeat (i outer)
    const float4* s4 = (const float4*)(S + b * 32);
    const float4* u4 = (const float4*)(U + (b * N_NODES + j) * 32);
    const float4* v4 = (const float4*)(V + (b * N_NODES + i) * 32);
    const float4* w4 = (const float4*)aw2;
    float acc = ab2[0];
#pragma unroll
    for (int q = 0; q < 8; ++q) {
        float4 ss = s4[q], us = u4[q], vs = v4[q], ws = w4[q];
        acc += fmaxf(ss.x + us.x + vs.x, 0.f) * ws.x;
        acc += fmaxf(ss.y + us.y + vs.y, 0.f) * ws.y;
        acc += fmaxf(ss.z + us.z + vs.z, 0.f) * ws.z;
        acc += fmaxf(ss.w + us.w + vs.w, 0.f) * ws.w;
    }
    out[idx] = acc;
}

// in-place softmax over 50625 logits; one 1024-thread block per batch element
__global__ void k_softmax(float* __restrict__ pi) {
    __shared__ float red[1024];
    float* l = pi + blockIdx.x * NPAIR;
    int t = threadIdx.x;
    float m = -1e30f;
    for (int p = t; p < NPAIR; p += 1024) m = fmaxf(m, l[p]);
    red[t] = m; __syncthreads();
    for (int s = 512; s > 0; s >>= 1) {
        if (t < s) red[t] = fmaxf(red[t], red[t + s]);
        __syncthreads();
    }
    float bm = red[0]; __syncthreads();
    float sum = 0.f;
    for (int p = t; p < NPAIR; p += 1024) sum += __expf(l[p] - bm);
    red[t] = sum; __syncthreads();
    for (int s = 512; s > 0; s >>= 1) {
        if (t < s) red[t] += red[t + s];
        __syncthreads();
    }
    float r = 1.0f / red[0];
    for (int p = t; p < NPAIR; p += 1024) l[p] = __expf(l[p] - bm) * r;
}

// ---------------------------------------------------------------- launcher
extern "C" void kernel_launch(void* const* d_in, const int* in_sizes, int n_in,
                              void* d_out, int out_size, void* d_ws, size_t ws_size,
                              hipStream_t stream) {
    const float* x        = (const float*)d_in[0];
    const int*   ei       = (const int*)d_in[1];   // [2, E]: row0 = src, row1 = dst
    const float* g0_w1    = (const float*)d_in[3];
    const float* g0_b1    = (const float*)d_in[4];
    const float* g0_gamma = (const float*)d_in[5];
    const float* g0_beta  = (const float*)d_in[6];
    const float* g0_w2    = (const float*)d_in[7];
    const float* g0_b2    = (const float*)d_in[8];
    const float* g1_w1    = (const float*)d_in[9];
    const float* g1_b1    = (const float*)d_in[10];
    const float* g1_gamma = (const float*)d_in[11];
    const float* g1_beta  = (const float*)d_in[12];
    const float* g1_w2    = (const float*)d_in[13];
    const float* g1_b2    = (const float*)d_in[14];
    const float* a_w1     = (const float*)d_in[15];  // [192,32]
    const float* a_b1     = (const float*)d_in[16];
    const float* a_w2     = (const float*)d_in[17];  // [32,1]
    const float* a_b2     = (const float*)d_in[18];
    const float* c_w1     = (const float*)d_in[19];  // [64,32]
    const float* c_b1     = (const float*)d_in[20];
    const float* c_w2     = (const float*)d_in[21];
    const float* c_b2     = (const float*)d_in[22];

    float* ws = (float*)d_ws;
    float* a0   = ws;                 // 3600
    float* bufA = ws + 3600;          // MPAD*64 = 115712  (h1, then h3)
    float* bufB = bufA + MPAD * HID;  // 115712  (h2, then feats)
    float* bufC = bufB + MPAD * HID;  // 115712  (a1)
    float* sc0  = bufC + MPAD * HID;  // 64
    float* sh0  = sc0 + 64;
    float* sc1  = sh0 + 64;
    float* sh1  = sc1 + 64;
    float* emb  = sh1 + 64;           // 512
    float* Sb   = emb + 512;          // 256
    float* U    = Sb + 256;           // 57600
    float* V    = U + 57600;          // 57600
    _Float16* bpW02 = (_Float16*)(V + 57600);   // 2048 halves (g0_w2, N=64)
    _Float16* bpW11 = bpW02 + 2048;             // 2048 (g1_w1)
    _Float16* bpW12 = bpW11 + 2048;             // 2048 (g1_w2)
    _Float16* bpU   = bpW12 + 2048;             // 1024 (a_w1 mid, N=32)
    _Float16* bpV   = bpU + 1024;               // 1024 (a_w1 last, N=32)

    float* pi  = (float*)d_out;                 // 8*50625
    float* val = (float*)d_out + BATCH * NPAIR; // 8

    const int mtiles = MPAD / 16;               // 113

    // ---- pack all WMMA B-operand weights once (tiny)
    k_pack_b<<<(4096 + 255) / 256, 256, 0, stream>>>(g0_w2, HID, HID, bpW02);
    k_pack_b<<<(4096 + 255) / 256, 256, 0, stream>>>(g1_w1, HID, HID, bpW11);
    k_pack_b<<<(4096 + 255) / 256, 256, 0, stream>>>(g1_w2, HID, HID, bpW12);
    k_pack_b<<<(2048 + 255) / 256, 256, 0, stream>>>(a_w1 + 64 * 32, 32, 32, bpU);
    k_pack_b<<<(2048 + 255) / 256, 256, 0, stream>>>(a_w1 + 128 * 32, 32, 32, bpV);

    // ---- GIN layer 0
    k_copy<<<(TOTAL * 2 + 255) / 256, 256, 0, stream>>>(x, a0, TOTAL * 2);
    k_edge_scatter<<<(E_CNT * 2 + 255) / 256, 256, 0, stream>>>(x, ei, ei + E_CNT, a0, 2);
    k_lin_in<<<(TOTAL * HID + 255) / 256, 256, 0, stream>>>(a0, g0_w1, g0_b1, bufA);
    k_bn_stats<<<HID, 256, 0, stream>>>(bufA, g0_gamma, g0_beta, sc0, sh0);
    k_gemm64_wmma<1><<<mtiles, 128, 0, stream>>>(bufA, TOTAL, sc0, sh0, bpW02,
                                                 g0_b2, bufB, HID);
    // ---- GIN layer 1
    k_copy<<<(TOTAL * HID + 255) / 256, 256, 0, stream>>>(bufB, bufC, TOTAL * HID);
    k_edge_scatter<<<(E_CNT * HID + 255) / 256, 256, 0, stream>>>(bufB, ei, ei + E_CNT, bufC, HID);
    k_gemm64_wmma<0><<<mtiles, 128, 0, stream>>>(bufC, TOTAL, nullptr, nullptr, bpW11,
                                                 g1_b1, bufA, HID);
    k_bn_stats<<<HID, 256, 0, stream>>>(bufA, g1_gamma, g1_beta, sc1, sh1);
    k_gemm64_wmma<1><<<mtiles, 128, 0, stream>>>(bufA, TOTAL, sc1, sh1, bpW12,
                                                 g1_b2, bufB, HID);
    // ---- pooling + critic
    k_pool<<<1, 512, 0, stream>>>(bufB, emb);
    k_critic<<<1, 32, 0, stream>>>(emb, c_w1, c_b1, c_w2, c_b2, val);
    // ---- actor (factored): S + U + V
    k_state_proj<<<1, 256, 0, stream>>>(emb, a_w1, a_b1, Sb);
    k_gemm64_wmma<0><<<mtiles, 64, 0, stream>>>(bufB, TOTAL, nullptr, nullptr, bpU,
                                                nullptr, U, 32);
    k_gemm64_wmma<0><<<mtiles, 64, 0, stream>>>(bufB, TOTAL, nullptr, nullptr, bpV,
                                                nullptr, V, 32);
    k_logits<<<(BATCH * NPAIR + 255) / 256, 256, 0, stream>>>(Sb, U, V, a_w2, a_b2, pi);
    k_softmax<<<BATCH, 1024, 0, stream>>>(pi);
}
